// Discriminator_79688823210569
// MI455X (gfx1250) — matrix operand order repaired and checked by reading, not compiled, not run on hardware
//
#include <hip/hip_runtime.h>
#include <hip/hip_bf16.h>
#include <stdint.h>

#define HIDDEN 64
#define DATA   16
#define C_CH   17      // 1 + DATA
#define T_LEN  64
#define B_ALL  1024
#define N3     (HIDDEN * C_CH)   // 1088
#define BT     32                // batch rows per workgroup
#define WROWS  (64 + 64 + N3)    // 1216 rows of 64 fp16 weights (W1t|W2t|W3t)

typedef __attribute__((ext_vector_type(16))) _Float16 v16h;
typedef __attribute__((ext_vector_type(8)))  _Float16 v8h;
typedef __attribute__((ext_vector_type(8)))  float    v8f;
typedef __attribute__((ext_vector_type(4)))  uint32_t u32x4;
typedef __attribute__((ext_vector_type(8)))  uint32_t u32x8;

static __device__ __forceinline__ v8f wmma_f16(v16h a, v16h b, v8f c) {
  return __builtin_amdgcn_wmma_f32_16x16x32_f16(
      /*neg_a=*/false, a, /*neg_b=*/false, b,
      /*c_mod=*/(short)0, c, /*reuse_a=*/false, /*reuse_b=*/false);
}

// Fast activations: v_exp_f32 + v_rcp_f32, no IEEE-division sequences.
static __device__ __forceinline__ float fast_sigmoid(float x) {
  return __builtin_amdgcn_rcpf(1.0f + __expf(-x));
}
static __device__ __forceinline__ float lipswish(float x) {
  return 0.909f * x * fast_sigmoid(x);
}
static __device__ __forceinline__ float fast_tanh(float x) {
  // tanh(x) = 1 - 2/(exp(2x)+1); saturates correctly at +/-inf
  float e = __expf(2.0f * x);
  return 1.0f - 2.0f * __builtin_amdgcn_rcpf(e + 1.0f);
}

// A fragment: 16x32 tile of row-major fp16 [m][k] with row stride `stride`.
// Lane L<16 holds row m0+L, K = k0..k0+15; L>=16 holds K = k0+16..k0+31.
static __device__ __forceinline__ v16h load_a(const _Float16* base, int stride,
                                              int m0, int k0, int lane) {
  int m  = m0 + (lane & 15);
  int hi = (lane >> 4) & 1;
  return *(const v16h*)(base + m * stride + k0 + 16 * hi);
}

// B fragment: 32x16 tile (KxN) from transposed weights Wt[n][k] (stride in halves).
// Lanes 0-15: K = k0..k0+15 ; lanes 16-31: K = k0+16..k0+31 ; N = lane&15.
static __device__ __forceinline__ v16h load_b(const _Float16* wt, int stride,
                                              int n0, int k0, int lane) {
  int n  = n0 + (lane & 15);
  int hi = (lane >> 4) & 1;
  return *(const v16h*)(wt + n * stride + k0 + 16 * hi);
}

// One 32x64 = (32x64 fp16) x (64x64 fp16, transposed) layer with bias+lipswish.
// Wave `wave` handles tile (m0 = (wave>>2)*16, n0 = (wave&3)*16).
static __device__ __forceinline__ void gemm_layer64(const _Float16* inA,
                                                    const _Float16* wt,
                                                    const float* bias,
                                                    _Float16* outA,
                                                    int lane, int wave) {
  int m0 = (wave >> 2) * 16;
  int n0 = (wave & 3) * 16;
  v8f c = {};
  c = wmma_f16(load_a(inA, HIDDEN, m0, 0, lane),  load_b(wt, HIDDEN, n0, 0, lane),  c);
  c = wmma_f16(load_a(inA, HIDDEN, m0, 32, lane), load_b(wt, HIDDEN, n0, 32, lane), c);
  int hi = lane >> 4;
  int n  = n0 + (lane & 15);
  float bn = bias[n];
#pragma unroll
  for (int r = 0; r < 8; ++r) {
    int m = m0 + 8 * hi + r;
    outA[m * HIDDEN + n] = (_Float16)lipswish(c[r] + bn);
  }
}

// Full vector-field evaluation f(t, z, dx) for the workgroup's 32 batch rows.
// Writes sF (32x64 fp32). Caller must __syncthreads() after.
static __device__ __forceinline__ void feval(
    float t, const float* fb1, const float* w1time,
    const _Float16* sW1, const _Float16* sW2, const _Float16* sW3,
    float* sB1, const float* sB2, const float* sB3,
    const _Float16* sZ, _Float16* sA1, _Float16* sA2, _Float16* sO,
    float* sF, const float* sDX, int tid) {
  // effective layer-1 bias (time channel folded in)
  for (int i = tid; i < HIDDEN; i += 256) sB1[i] = fb1[i] + t * w1time[i];
  __syncthreads();

  int lane = tid & 31, wave = tid >> 5;

  gemm_layer64(sZ, sW1, sB1, sA1, lane, wave);
  __syncthreads();
  gemm_layer64(sA1, sW2, sB2, sA2, lane, wave);
  __syncthreads();

  // layer 3 phase A: tanh(act2 @ W3 + b3) -> sO (32x1088 fp16), no atomics
#pragma unroll
  for (int tm = 0; tm < 2; ++tm) {
    v16h a0 = load_a(sA2, HIDDEN, tm * 16, 0, lane);
    v16h a1 = load_a(sA2, HIDDEN, tm * 16, 32, lane);
    for (int tn = wave; tn < N3 / 16; tn += 8) {     // 68 N-tiles over 8 waves
      int n0 = tn * 16;
      v8f c = {};
      c = wmma_f16(a0, load_b(sW3, HIDDEN, n0, 0, lane),  c);
      c = wmma_f16(a1, load_b(sW3, HIDDEN, n0, 32, lane), c);
      int hi = lane >> 4;
      int ng = n0 + (lane & 15);          // output column in [0,1088)
      float bn = sB3[ng];
#pragma unroll
      for (int r = 0; r < 8; ++r) {
        int m = tm * 16 + 8 * hi + r;
        sO[m * N3 + ng] = (_Float16)fast_tanh(c[r] + bn);
      }
    }
  }
  __syncthreads();

  // phase B: f[m][h] = sum_c O[m][h*17+c] * dx[m][c]  (private dot, no RMW)
  for (int i = tid; i < BT * HIDDEN; i += 256) {
    int m = i >> 6, h = i & 63;
    const _Float16* row = sO + m * N3 + h * C_CH;
    const float*    dxr = sDX + m * C_CH;
    float acc = 0.0f;
#pragma unroll
    for (int c = 0; c < C_CH; ++c) acc += (float)row[c] * dxr[c];
    sF[i] = acc;
  }
}

// ---------------------------------------------------------------------------
// Kernel 1: convert func-MLP weights to fp16, transposed [n][k] layout.
__global__ __launch_bounds__(256) void prep_weights(
    const float* fW1, const float* fW2, const float* fW3,
    float* w1time, _Float16* W1t, _Float16* W2t, _Float16* W3t) {
  int i = blockIdx.x * blockDim.x + threadIdx.x;
  int stride = gridDim.x * blockDim.x;
  for (int idx = i; idx < HIDDEN; idx += stride)
    w1time[idx] = fW1[idx];                               // row 0 of 65x64 (time)
  for (int idx = i; idx < HIDDEN * HIDDEN; idx += stride) {
    int n = idx >> 6, k = idx & 63;
    W1t[idx] = (_Float16)fW1[(k + 1) * HIDDEN + n];       // skip time row
  }
  for (int idx = i; idx < HIDDEN * HIDDEN; idx += stride) {
    int n = idx >> 6, k = idx & 63;
    W2t[idx] = (_Float16)fW2[k * HIDDEN + n];
  }
  for (int idx = i; idx < N3 * HIDDEN; idx += stride) {
    int n = idx >> 6, k = idx & 63;
    W3t[idx] = (_Float16)fW3[k * N3 + n];
  }
}

// Kernel 2: h0 = FFNN_init(ys[:,0]) with final sigmoid. One block per batch row.
__global__ __launch_bounds__(64) void init_h0(
    const float* ys, const float* iW1, const float* ib1, const float* iW2,
    const float* ib2, const float* iW3, const float* ib3, float* h0) {
  __shared__ float sx[C_CH];
  __shared__ float sa[HIDDEN];
  __shared__ float sb[HIDDEN];
  int b = blockIdx.x, j = threadIdx.x;
  if (j < C_CH) sx[j] = ys[(size_t)b * T_LEN * C_CH + j];
  __syncthreads();
  float s = ib1[j];
  for (int i = 0; i < C_CH; ++i) s += sx[i] * iW1[i * HIDDEN + j];
  sa[j] = lipswish(s);
  __syncthreads();
  s = ib2[j];
  for (int i = 0; i < HIDDEN; ++i) s += sa[i] * iW2[i * HIDDEN + j];
  sb[j] = lipswish(s);
  __syncthreads();
  s = ib3[j];
  for (int i = 0; i < HIDDEN; ++i) s += sb[i] * iW3[i * HIDDEN + j];
  h0[(size_t)b * HIDDEN + j] = fast_sigmoid(s);
}

// Kernel 3: the full 63-step reversible-Heun scan, weights resident in LDS.
// Weight stage-in uses the Tensor Data Mover (one D# moving 1216x64 fp16).
__global__ __launch_bounds__(256) void scan_kernel(
    const float* ys, const float* fb1, const float* fb2, const float* fb3,
    const float* w1time, const _Float16* W1t, const _Float16* W2t,
    const _Float16* W3t, float* hbuf /* h0 in, yT out */) {
  extern __shared__ char smem_raw[];
  _Float16* sW1 = (_Float16*)smem_raw;          // 64x64    (LDS offset 0)
  _Float16* sW2 = sW1 + HIDDEN * HIDDEN;        // 64x64
  _Float16* sW3 = sW2 + HIDDEN * HIDDEN;        // 1088x64
  _Float16* sZ  = sW3 + N3 * HIDDEN;            // 32x64 fp16 (WMMA A input)
  _Float16* sA1 = sZ  + BT * HIDDEN;
  _Float16* sA2 = sA1 + BT * HIDDEN;
  _Float16* sO  = sA2 + BT * HIDDEN;            // 32x1088 fp16 tanh(out3)
  float* sB1  = (float*)(sO + BT * N3);         // 64
  float* sB2  = sB1 + HIDDEN;                   // 64
  float* sB3  = sB2 + HIDDEN;                   // 1088
  float* sY   = sB3 + N3;                       // 32x64 fp32
  float* sZf  = sY  + BT * HIDDEN;              // 32x64 fp32
  float* sF   = sZf + BT * HIDDEN;              // 32x64 fp32
  float* sDX0 = sF  + BT * HIDDEN;              // 32x17
  float* sDX1 = sDX0 + BT * C_CH;               // 32x17

  int tid  = threadIdx.x;
  int wave = tid >> 5;
  int b0   = blockIdx.x * BT;

  // --- TDM: DMA the whole contiguous fp16 weight block (W1t|W2t|W3t) into
  // LDS offset 0 as a 2D tile: 1216 rows x 64 halves, stride 64, 2B elements.
  if (wave == 0) {
    uint64_t gaddr = (uint64_t)(uintptr_t)W1t;           // 57-bit global byte addr
    u32x4 g0;
    g0[0] = 1u;                                          // count=1, user descriptor
    g0[1] = 0u;                                          // lds_addr = 0 (sW1)
    g0[2] = (uint32_t)gaddr;                             // global_addr[31:0]
    g0[3] = (uint32_t)((gaddr >> 32) & 0x01FFFFFFu)      // global_addr[56:32]
            | (2u << 30);                                // type = 2 ("image")
    u32x8 g1;
    g1[0] = 1u << 16;                                    // data_size=1 (2 bytes)
    g1[1] = (uint32_t)HIDDEN << 16;                      // tensor_dim0[15:0] = 64
    g1[2] = (uint32_t)WROWS << 16;                       // dim0 hi=0 | tensor_dim1 lo=1216
    g1[3] = (uint32_t)HIDDEN << 16;                      // dim1 hi=0 | tile_dim0 = 64
    g1[4] = (uint32_t)WROWS;                             // tile_dim1 = 1216, tile_dim2=0
    g1[5] = (uint32_t)HIDDEN;                            // tensor_dim0_stride lo = 64
    g1[6] = 0u;                                          // stride0 hi | dim1_stride lo
    g1[7] = 0u;                                          // dim1_stride hi
    asm volatile("tensor_load_to_lds %0, %1" :: "s"(g0), "s"(g1) : "memory");
  }

  // --- overlap: biases + state init while the TDM transfer is in flight ---
  for (int i = tid; i < HIDDEN; i += 256) sB2[i] = fb2[i];
  for (int i = tid; i < N3; i += 256)     sB3[i] = fb3[i];
  for (int i = tid; i < BT * HIDDEN; i += 256) {
    float v = hbuf[(size_t)(b0 + (i >> 6)) * HIDDEN + (i & 63)];
    sY[i] = v; sZf[i] = v; sZ[i] = (_Float16)v;
  }
  if (wave == 0) __builtin_amdgcn_s_wait_tensorcnt(0);
  __syncthreads();

  for (int n = 0; n < T_LEN - 1; ++n) {
    // dx at t_n and t_{n+1} (last derivative repeated)
    int n2 = (n + 1 < T_LEN - 2) ? n + 1 : T_LEN - 2;
    for (int i = tid; i < BT * C_CH; i += 256) {
      int bl = i / C_CH, c = i - bl * C_CH;
      const float* base = ys + (size_t)(b0 + bl) * T_LEN * C_CH;
      sDX0[i] = base[(n + 1) * C_CH + c] - base[n * C_CH + c];
      sDX1[i] = base[(n2 + 1) * C_CH + c] - base[n2 * C_CH + c];
    }
    __syncthreads();

    // f0 = f(t_n, z, dx0)
    feval((float)n, fb1, w1time, sW1, sW2, sW3, sB1, sB2, sB3,
          sZ, sA1, sA2, sO, sF, sDX0, tid);
    __syncthreads();

    // z1 = 2y - z + f0 ; y += 0.5 f0
    for (int i = tid; i < BT * HIDDEN; i += 256) {
      float y = sY[i], z = sZf[i], f0 = sF[i];
      float z1 = 2.0f * y - z + f0;
      sY[i]  = y + 0.5f * f0;
      sZf[i] = z1;
      sZ[i]  = (_Float16)z1;
    }
    __syncthreads();

    // f1 = f(t_n + 1, z1, dx1)
    feval((float)(n + 1), fb1, w1time, sW1, sW2, sW3, sB1, sB2, sB3,
          sZ, sA1, sA2, sO, sF, sDX1, tid);
    __syncthreads();

    for (int i = tid; i < BT * HIDDEN; i += 256) sY[i] += 0.5f * sF[i];
    __syncthreads();
  }

  // write yT back over the h0 scratch region
  for (int i = tid; i < BT * HIDDEN; i += 256)
    hbuf[(size_t)(b0 + (i >> 6)) * HIDDEN + (i & 63)] = sY[i];
}

// Kernel 4: mean(yT @ rW + rb)
__global__ __launch_bounds__(256) void readout(const float* yT, const float* rW,
                                               const float* rb, float* out) {
  __shared__ float red[256];
  int tid = threadIdx.x;
  float acc = 0.0f;
  for (int b = tid; b < B_ALL; b += 256) {
    float s = rb[0];
    for (int h = 0; h < HIDDEN; ++h) s += yT[(size_t)b * HIDDEN + h] * rW[h];
    acc += s;
  }
  red[tid] = acc;
  __syncthreads();
  for (int off = 128; off > 0; off >>= 1) {
    if (tid < off) red[tid] += red[tid + off];
    __syncthreads();
  }
  if (tid == 0) out[0] = red[0] / (float)B_ALL;
}

extern "C" void kernel_launch(void* const* d_in, const int* in_sizes, int n_in,
                              void* d_out, int out_size, void* d_ws, size_t ws_size,
                              hipStream_t stream) {
  (void)in_sizes; (void)n_in; (void)out_size; (void)ws_size;
  // dict order: ys_coeffs, init (W1,b1,W2,b2,W3,b3), func (W1,b1,W2,b2,W3,b3), rW, rb
  const float* ys  = (const float*)d_in[0];
  const float* iW1 = (const float*)d_in[1];
  const float* ib1 = (const float*)d_in[2];
  const float* iW2 = (const float*)d_in[3];
  const float* ib2 = (const float*)d_in[4];
  const float* iW3 = (const float*)d_in[5];
  const float* ib3 = (const float*)d_in[6];
  const float* fW1 = (const float*)d_in[7];
  const float* fb1 = (const float*)d_in[8];
  const float* fW2 = (const float*)d_in[9];
  const float* fb2 = (const float*)d_in[10];
  const float* fW3 = (const float*)d_in[11];
  const float* fb3 = (const float*)d_in[12];
  const float* rW  = (const float*)d_in[13];
  const float* rb  = (const float*)d_in[14];

  // workspace carve
  float* hbuf     = (float*)d_ws;                       // 1024*64 fp32 (h0 / yT)
  float* w1time   = hbuf + B_ALL * HIDDEN;              // 64 fp32 (pad to 128)
  _Float16* W1t   = (_Float16*)(w1time + 128);          // 64x64 fp16 transposed
  _Float16* W2t   = W1t + HIDDEN * HIDDEN;              // contiguous with W1t
  _Float16* W3t   = W2t + HIDDEN * HIDDEN;              // 1088x64, contiguous

  prep_weights<<<64, 256, 0, stream>>>(fW1, fW2, fW3, w1time, W1t, W2t, W3t);
  init_h0<<<B_ALL, 64, 0, stream>>>(ys, iW1, ib1, iW2, ib2, iW3, ib3, hbuf);

  size_t smem =
      sizeof(_Float16) * (2 * HIDDEN * HIDDEN + N3 * HIDDEN +
                          3 * BT * HIDDEN + BT * N3) +
      sizeof(float) * (2 * HIDDEN + N3 + 3 * BT * HIDDEN + 2 * BT * C_CH);
  scan_kernel<<<B_ALL / BT, 256, smem, stream>>>(ys, fb1, fb2, fb3, w1time,
                                                 W1t, W2t, W3t, hbuf);
  readout<<<1, 256, 0, stream>>>(hbuf, rW, rb, (float*)d_out);
}
